// TransformerLayerB_18262200942698
// MI455X (gfx1250) — compile-verified, matrix operand order checked by reading, not run on hardware
//
#include <hip/hip_runtime.h>

// ---------------------------------------------------------------------------
// MI455X (gfx1250) transformer layer: f16 WMMA GEMMs + flash attention.
// ~164 GFLOP total vs ~9us of HBM traffic at 23.3 TB/s -> compute bound, so
// all matmuls run on v_wmma_f32_16x16x32_f16 (f32 accum) and the TxT score
// matrix is never materialized (online softmax). GEMMs use BK=64 double-
// buffered LDS with one barrier per K-step; tile fill uses the CDNA5 async
// global->LDS path (ASYNCcnt) when the toolchain exposes it.
// ---------------------------------------------------------------------------

typedef __attribute__((ext_vector_type(16))) _Float16 v16h_t;
typedef __attribute__((ext_vector_type(8)))  _Float16 v8h_t;
typedef __attribute__((ext_vector_type(8)))  float    v8f_t;
typedef __attribute__((ext_vector_type(4)))  int      v4i_t;

// Problem constants (match reference)
constexpr int NB    = 2;
constexpr int T_SEQ = 2048;
constexpr int DKQ   = 1024;
constexpr int EKQ   = 1024;
constexpr int HH    = 16;
constexpr int HD    = 64;          // head dim
constexpr int FF    = 4096;        // 4*E
constexpr int MROWS = NB * T_SEQ;  // 4096

// --- optional CDNA5 async global->LDS path (ASYNCcnt-tracked, ISA §15.18) ---
#if __has_builtin(__builtin_amdgcn_global_load_async_to_lds_b128)
#define USE_ASYNC_LDS 1
typedef __attribute__((address_space(1))) v4i_t* gv4_t;  // global int4*
typedef __attribute__((address_space(3))) v4i_t* lv4_t;  // LDS int4*
__device__ __forceinline__ void async_cp16(const void* g, void* l) {
  __builtin_amdgcn_global_load_async_to_lds_b128((gv4_t)g, (lv4_t)l, 0, 0);
}
__device__ __forceinline__ void async_wait0() {
#if __has_builtin(__builtin_amdgcn_s_wait_asynccnt)
  __builtin_amdgcn_s_wait_asynccnt(0);
#else
  asm volatile("s_wait_asynccnt 0x0" ::: "memory");
#endif
}
#endif

__device__ __forceinline__ v8f_t zero8() {
  v8f_t z = {0.f, 0.f, 0.f, 0.f, 0.f, 0.f, 0.f, 0.f};
  return z;
}

__device__ __forceinline__ v8f_t wmma16(v16h_t a, v16h_t b, v8f_t c) {
  // D = A(16x32) * B(32x16) + C, f32 accum
  return __builtin_amdgcn_wmma_f32_16x16x32_f16(false, a, false, b, (short)0, c,
                                                false, false);
}

// Load one 16x32 f16 A/B fragment from a row-major tile (row stride `ld`
// halves). CDNA5 16-bit A/B layout: lanes 0-15 hold K={0..7,16..23}, lanes
// 16-31 hold K={8..15,24..31} of row (lane&15): two 16B chunks at +0,+32B
// (+16B for the upper half-wave).
__device__ __forceinline__ v16h_t load_frag(const _Float16* tile, int ld, int lane) {
  const _Float16* p = tile + (size_t)(lane & 15) * ld + ((lane >> 4) & 1) * 8;
  union { v16h_t v; v8h_t h[2]; } u;
  u.h[0] = *(const v8h_t*)(p);
  u.h[1] = *(const v8h_t*)(p + 16);
  return u.v;
}

// ---------------------------------------------------------------------------
// One-time precision conversion kernels
// ---------------------------------------------------------------------------
__global__ void k_cvt_f16(const float* __restrict__ in, _Float16* __restrict__ out,
                          int n) {
  int i = blockIdx.x * 256 + threadIdx.x;
  if (i < n) out[i] = (_Float16)in[i];
}

// in: [K][N] f32 row-major -> out: [N][K] f16 (transposed weights so B-frags
// are contiguous in K)
__global__ void k_trans_f16(const float* __restrict__ in, _Float16* __restrict__ out,
                            int K, int N) {
  int i = blockIdx.x * 256 + threadIdx.x;
  if (i < K * N) {
    int k = i / N, n = i % N;
    out[(size_t)n * K + k] = (_Float16)in[i];
  }
}

// ---------------------------------------------------------------------------
// Tiled WMMA GEMM:  C[M,N] = A[M,K](f16) * Bt[N,K](f16)^T + bias
// Block tile 128x128, 4 waves (2x2), each wave 64x64 = 16 accum tiles.
// BK=64, double-buffered LDS, one barrier per K-step, async or pipelined fill.
// EPI: 0 = f16 store, 1 = exact-GELU f16 store, 2 = f32 store.
// ---------------------------------------------------------------------------
template <int EPI>
__global__ __launch_bounds__(128) void k_gemm(
    const _Float16* __restrict__ A, const _Float16* __restrict__ Bt,
    const float* __restrict__ bias, _Float16* __restrict__ Oh,
    float* __restrict__ Of, int M, int N, int K) {
  __shared__ _Float16 As[2][128 * 64];
  __shared__ _Float16 Bs[2][128 * 64];

  const int t    = threadIdx.x;
  const int lane = t & 31;
  const int w    = t >> 5;
  const int wr   = (w >> 1) * 64;
  const int wc   = (w & 1) * 64;
  const int bm   = blockIdx.y * 128;
  const int bn   = blockIdx.x * 128;
  const int NS   = K / 64;

  v8f_t acc[4][4];
#pragma unroll
  for (int i = 0; i < 4; ++i)
#pragma unroll
    for (int j = 0; j < 4; ++j) acc[i][j] = zero8();

  // Each thread owns one A row and one Bt row of the block tile.
  const _Float16* arow = A + (size_t)(bm + t) * K;
  const _Float16* brow = Bt + (size_t)(bn + t) * K;

#ifdef USE_ASYNC_LDS
  // ---- async global->LDS double-buffered pipeline ----
  {
#pragma unroll
    for (int i = 0; i < 8; ++i) {
      async_cp16(arow + i * 8, &As[0][t * 64 + i * 8]);
      async_cp16(brow + i * 8, &Bs[0][t * 64 + i * 8]);
    }
  }
  async_wait0();
  __syncthreads();
  for (int ks = 0; ks < NS; ++ks) {
    const int cur = ks & 1;
    if (ks + 1 < NS) {
      const _Float16* ag = arow + (ks + 1) * 64;
      const _Float16* bg = brow + (ks + 1) * 64;
#pragma unroll
      for (int i = 0; i < 8; ++i) {
        async_cp16(ag + i * 8, &As[1 - cur][t * 64 + i * 8]);
        async_cp16(bg + i * 8, &Bs[1 - cur][t * 64 + i * 8]);
      }
      if (ks + 2 < NS) {
        __builtin_prefetch(arow + (ks + 2) * 64, 0, 1);
        __builtin_prefetch(brow + (ks + 2) * 64, 0, 1);
      }
    }
#pragma unroll
    for (int kc = 0; kc < 2; ++kc) {
      v16h_t af[4], bf[4];
#pragma unroll
      for (int i = 0; i < 4; ++i)
        af[i] = load_frag(&As[cur][(wr + 16 * i) * 64 + kc * 32], 64, lane);
#pragma unroll
      for (int j = 0; j < 4; ++j)
        bf[j] = load_frag(&Bs[cur][(wc + 16 * j) * 64 + kc * 32], 64, lane);
#pragma unroll
      for (int i = 0; i < 4; ++i)
#pragma unroll
        for (int j = 0; j < 4; ++j) acc[i][j] = wmma16(af[i], bf[j], acc[i][j]);
    }
    async_wait0();
    __syncthreads();
  }
#else
  // ---- register-staged double-buffered pipeline (one barrier per K-step) ----
  uint4 ra[8], rb[8];
  {
    const uint4* ag = (const uint4*)(arow);
    const uint4* bg = (const uint4*)(brow);
#pragma unroll
    for (int i = 0; i < 8; ++i) { ra[i] = ag[i]; rb[i] = bg[i]; }
  }
  for (int ks = 0; ks < NS; ++ks) {
    const int cur = ks & 1;
    {
      uint4* ad = (uint4*)&As[cur][t * 64];
      uint4* bd = (uint4*)&Bs[cur][t * 64];
#pragma unroll
      for (int i = 0; i < 8; ++i) { ad[i] = ra[i]; bd[i] = rb[i]; }
    }
    if (ks + 1 < NS) {
      const uint4* ag = (const uint4*)(arow + (ks + 1) * 64);
      const uint4* bg = (const uint4*)(brow + (ks + 1) * 64);
#pragma unroll
      for (int i = 0; i < 8; ++i) { ra[i] = ag[i]; rb[i] = bg[i]; }
      if (ks + 2 < NS) {
        __builtin_prefetch(arow + (ks + 2) * 64, 0, 1);
        __builtin_prefetch(brow + (ks + 2) * 64, 0, 1);
      }
    }
    __syncthreads();  // buf[cur] ready; WAR on buf[1-cur] covered by prev barrier
#pragma unroll
    for (int kc = 0; kc < 2; ++kc) {
      v16h_t af[4], bf[4];
#pragma unroll
      for (int i = 0; i < 4; ++i)
        af[i] = load_frag(&As[cur][(wr + 16 * i) * 64 + kc * 32], 64, lane);
#pragma unroll
      for (int j = 0; j < 4; ++j)
        bf[j] = load_frag(&Bs[cur][(wc + 16 * j) * 64 + kc * 32], 64, lane);
#pragma unroll
      for (int i = 0; i < 4; ++i)
#pragma unroll
        for (int j = 0; j < 4; ++j) acc[i][j] = wmma16(af[i], bf[j], acc[i][j]);
    }
  }
#endif

  // Epilogue. C/D layout: VGPR v -> rows v / v+8 (lane halves); col = lane&15.
  const int hi8  = ((lane >> 4) & 1) * 8;
  const int ln15 = lane & 15;
#pragma unroll
  for (int i = 0; i < 4; ++i) {
#pragma unroll
    for (int j = 0; j < 4; ++j) {
      int col  = bn + wc + 16 * j + ln15;
      float bc = bias[col];
#pragma unroll
      for (int vv = 0; vv < 8; ++vv) {
        int row = bm + wr + 16 * i + vv + hi8;
        float x = acc[i][j][vv] + bc;
        if (EPI == 1) x = 0.5f * x * (1.0f + erff(x * 0.70710678118654752f));
        if (EPI == 2)
          Of[(size_t)row * N + col] = x;
        else
          Oh[(size_t)row * N + col] = (_Float16)x;
      }
    }
  }
}

// ---------------------------------------------------------------------------
// Flash attention, shifted-causal mask (key < query, plus [0,0]).
// Grid: (T/64, H, NB); block = 128 threads = 4 waves; wave w owns query rows
// [q0+16w, q0+16w+16). K/V 32-key blocks staged in LDS cooperatively.
// ---------------------------------------------------------------------------
__global__ __launch_bounds__(128) void k_attn(
    const _Float16* __restrict__ Q, const _Float16* __restrict__ Kp,
    const _Float16* __restrict__ Vp, _Float16* __restrict__ O) {
  __shared__ _Float16 Ks[32 * 64];      // [key][feature]
  __shared__ _Float16 Vt[64 * 32];      // [dim][key] (transposed for B-frags)
  __shared__ _Float16 Pb[4 * 16 * 32];  // per-wave P repack buffer

  const int t     = threadIdx.x;
  const int lane  = t & 31;
  const int w     = t >> 5;
  const int q0    = blockIdx.x * 64;
  const int h     = blockIdx.y;
  const int n     = blockIdx.z;
  const int qbase = q0 + w * 16;
  const int hi8   = ((lane >> 4) & 1) * 8;
  const int ln15  = lane & 15;

  // Q fragments: rows qbase..qbase+15, features 0..63 (two 16x32 A-frags)
  const _Float16* qrow = Q + ((size_t)(n * T_SEQ + qbase)) * EKQ + h * HD;
  v16h_t aq0 = load_frag(qrow, EKQ, lane);
  v16h_t aq1 = load_frag(qrow + 32, EKQ, lane);

  v8f_t o0 = zero8(), o1 = zero8(), o2 = zero8(), o3 = zero8();
  float mrow[8], lrow[8];
#pragma unroll
  for (int vv = 0; vv < 8; ++vv) {
    mrow[vv] = -__builtin_inff();
    lrow[vv] = 0.f;
  }

  const int nkb = q0 / 32 + 2;  // keys needed: up to q0+63 (strictly past)
  for (int kb = 0; kb < nkb; ++kb) {
    const int kbase = kb * 32;
    __syncthreads();
    {  // cooperative staging: thread -> (key, 16-half segment)
      int key = t >> 2, seg = t & 3;
      const _Float16* ksrc =
          Kp + ((size_t)(n * T_SEQ + kbase + key)) * EKQ + h * HD + seg * 16;
      uint4* kdst = (uint4*)(Ks + key * 64 + seg * 16);
      kdst[0] = ((const uint4*)ksrc)[0];
      kdst[1] = ((const uint4*)ksrc)[1];
      const _Float16* vsrc =
          Vp + ((size_t)(n * T_SEQ + kbase + key)) * EKQ + h * HD + seg * 16;
#pragma unroll
      for (int d = 0; d < 16; ++d) Vt[(seg * 16 + d) * 32 + key] = vsrc[d];
      if (kb + 1 < nkb) {
        __builtin_prefetch(ksrc + 32 * EKQ, 0, 1);
        __builtin_prefetch(vsrc + 32 * EKQ, 0, 1);
      }
    }
    __syncthreads();

    // S = Q * K^T : two 16x16 score tiles (keys kbase..+15, kbase+16..+31)
    v16h_t b00 = load_frag(Ks + 0, 64, lane);
    v16h_t b01 = load_frag(Ks + 32, 64, lane);
    v16h_t b10 = load_frag(Ks + 16 * 64, 64, lane);
    v16h_t b11 = load_frag(Ks + 16 * 64 + 32, 64, lane);
    v8f_t s0 = zero8(), s1 = zero8();
    s0 = wmma16(aq0, b00, s0);
    s0 = wmma16(aq1, b01, s0);
    s1 = wmma16(aq0, b10, s1);
    s1 = wmma16(aq1, b11, s1);

    // Online softmax update (row reductions via shfl over 16-lane halves)
    float p0[8], p1[8], sc[8];
#pragma unroll
    for (int vv = 0; vv < 8; ++vv) {
      int qr  = qbase + vv + hi8;
      int kc0 = kbase + ln15;
      int kc1 = kc0 + 16;
      float x0 = ((kc0 < qr) || (qr == 0 && kc0 == 0)) ? (float)s0[vv]
                                                       : -__builtin_inff();
      float x1 = (kc1 < qr) ? (float)s1[vv] : -__builtin_inff();
      float mx = fmaxf(x0, x1);
#pragma unroll
      for (int off = 1; off <= 8; off <<= 1)
        mx = fmaxf(mx, __shfl_xor(mx, off, 32));
      float mn = fmaxf(mrow[vv], mx);
      float s  = __expf(mrow[vv] - mn);
      float e0 = __expf(x0 - mn);
      float e1 = __expf(x1 - mn);
      float rs = e0 + e1;
#pragma unroll
      for (int off = 1; off <= 8; off <<= 1) rs += __shfl_xor(rs, off, 32);
      lrow[vv] = lrow[vv] * s + rs;
      mrow[vv] = mn;
      sc[vv] = s;
      p0[vv] = e0;
      p1[vv] = e1;
    }
#pragma unroll
    for (int vv = 0; vv < 8; ++vv) {
      o0[vv] *= sc[vv];
      o1[vv] *= sc[vv];
      o2[vv] *= sc[vv];
      o3[vv] *= sc[vv];
    }

    // Repack P from C/D layout to A layout via per-wave LDS buffer
    _Float16* pw = Pb + w * 512;
#pragma unroll
    for (int vv = 0; vv < 8; ++vv) {
      int rl = vv + hi8;
      pw[rl * 32 + ln15]      = (_Float16)p0[vv];
      pw[rl * 32 + 16 + ln15] = (_Float16)p1[vv];
    }
    v16h_t pa = load_frag(pw, 32, lane);
    v16h_t v0 = load_frag(Vt + 0, 32, lane);
    v16h_t v1 = load_frag(Vt + 16 * 32, 32, lane);
    v16h_t v2 = load_frag(Vt + 32 * 32, 32, lane);
    v16h_t v3 = load_frag(Vt + 48 * 32, 32, lane);
    o0 = wmma16(pa, v0, o0);
    o1 = wmma16(pa, v1, o1);
    o2 = wmma16(pa, v2, o2);
    o3 = wmma16(pa, v3, o3);
  }

  // Normalize by row sum and store f16
#pragma unroll
  for (int vv = 0; vv < 8; ++vv) {
    int qr    = qbase + vv + hi8;
    float inv = 1.f / lrow[vv];
    _Float16* orow = O + ((size_t)(n * T_SEQ + qr)) * EKQ + h * HD;
    orow[ln15]      = (_Float16)(o0[vv] * inv);
    orow[16 + ln15] = (_Float16)(o1[vv] * inv);
    orow[32 + ln15] = (_Float16)(o2[vv] * inv);
    orow[48 + ln15] = (_Float16)(o3[vv] * inv);
  }
}

// ---------------------------------------------------------------------------
// Host launcher
// ---------------------------------------------------------------------------
extern "C" void kernel_launch(void* const* d_in, const int* in_sizes, int n_in,
                              void* d_out, int out_size, void* d_ws, size_t ws_size,
                              hipStream_t stream) {
  (void)in_sizes; (void)n_in; (void)out_size; (void)ws_size;
  const float* kq = (const float*)d_in[0];
  const float* v  = (const float*)d_in[1];
  const float* Wk = (const float*)d_in[2];
  const float* bk = (const float*)d_in[3];
  const float* Wq = (const float*)d_in[4];
  const float* bq = (const float*)d_in[5];
  const float* Wv = (const float*)d_in[6];
  const float* bv = (const float*)d_in[7];
  const float* W1 = (const float*)d_in[8];
  const float* b1 = (const float*)d_in[9];
  const float* W2 = (const float*)d_in[10];
  const float* b2 = (const float*)d_in[11];
  float* out = (float*)d_out;

  char* ws   = (char*)d_ws;
  size_t off = 0;
  auto halves = [&](size_t nelem) -> _Float16* {
    _Float16* p = (_Float16*)(ws + off);
    off += ((nelem * 2) + 255) & ~(size_t)255;
    return p;
  };

  _Float16* Xh  = halves((size_t)MROWS * DKQ);  // kq in f16
  _Float16* Vih = halves((size_t)MROWS * DKQ);  // v in f16
  _Float16* WkT = halves((size_t)EKQ * DKQ);
  _Float16* WqT = halves((size_t)EKQ * DKQ);
  _Float16* WvT = halves((size_t)EKQ * DKQ);
  _Float16* W1T = halves((size_t)FF * EKQ);
  _Float16* W2T = halves((size_t)EKQ * FF);
  _Float16* Qh  = halves((size_t)MROWS * EKQ);
  _Float16* Kh  = halves((size_t)MROWS * EKQ);
  _Float16* Vh  = halves((size_t)MROWS * EKQ);
  _Float16* Oh  = halves((size_t)MROWS * EKQ);  // attention output
  _Float16* Hh  = halves((size_t)MROWS * FF);   // FFN hidden

  // 1) convert activations, transpose+convert weights
  {
    int n1 = MROWS * DKQ;
    k_cvt_f16<<<dim3((n1 + 255) / 256), 256, 0, stream>>>(kq, Xh, n1);
    k_cvt_f16<<<dim3((n1 + 255) / 256), 256, 0, stream>>>(v, Vih, n1);
    int nw = DKQ * EKQ;
    k_trans_f16<<<dim3((nw + 255) / 256), 256, 0, stream>>>(Wk, WkT, DKQ, EKQ);
    k_trans_f16<<<dim3((nw + 255) / 256), 256, 0, stream>>>(Wq, WqT, DKQ, EKQ);
    k_trans_f16<<<dim3((nw + 255) / 256), 256, 0, stream>>>(Wv, WvT, DKQ, EKQ);
    int nf = EKQ * FF;
    k_trans_f16<<<dim3((nf + 255) / 256), 256, 0, stream>>>(W1, W1T, EKQ, FF);
    k_trans_f16<<<dim3((nf + 255) / 256), 256, 0, stream>>>(W2, W2T, FF, EKQ);
  }

  // 2) K/Q/V projections (f16 out + bias)
  {
    dim3 g(EKQ / 128, MROWS / 128);
    k_gemm<0><<<g, 128, 0, stream>>>(Xh, WkT, bk, Kh, nullptr, MROWS, EKQ, DKQ);
    k_gemm<0><<<g, 128, 0, stream>>>(Xh, WqT, bq, Qh, nullptr, MROWS, EKQ, DKQ);
    k_gemm<0><<<g, 128, 0, stream>>>(Vih, WvT, bv, Vh, nullptr, MROWS, EKQ, DKQ);
  }

  // 3) flash attention
  k_attn<<<dim3(T_SEQ / 64, HH, NB), 128, 0, stream>>>(Qh, Kh, Vh, Oh);

  // 4) FFN: GELU(Oh*W1 + b1) * W2 + b2 -> f32 out
  {
    dim3 g1(FF / 128, MROWS / 128);
    k_gemm<1><<<g1, 128, 0, stream>>>(Oh, W1T, b1, Hh, nullptr, MROWS, FF, EKQ);
    dim3 g2(EKQ / 128, MROWS / 128);
    k_gemm<2><<<g2, 128, 0, stream>>>(Hh, W2T, b2, nullptr, out, MROWS, EKQ, FF);
  }
}
